// SpatialGCN_38757784879109
// MI455X (gfx1250) — matrix operand order repaired and checked by reading, not compiled, not run on hardware
//
#include <hip/hip_runtime.h>

// ---------------------------------------------------------------------------
// SpatialGCN on gfx1250: every GEMM via v_wmma_f32_16x16x32_bf16 (wave32),
// A-tiles staged with GLOBAL_LOAD_ASYNC_TO_LDS_B128 (ASYNCcnt path,
// CDNA5 ISA), B-tiles register-prefetched one K-step
// ahead, double-buffered LDS, one barrier per K-step.
// ---------------------------------------------------------------------------

typedef __attribute__((ext_vector_type(16))) __bf16 v16bf;
typedef __attribute__((ext_vector_type(8)))  __bf16 v8bf;
typedef __attribute__((ext_vector_type(8)))  float  v8f;

#define EP_BIAS    1
#define EP_BN      2
#define EP_TRANS   4
#define EP_F32OUT  8
#define EP_ADDRELU 16

__global__ __launch_bounds__(256)
void sgcn_cvt_f32_bf16(const float* __restrict__ in, __bf16* __restrict__ out, int n) {
  int i = blockIdx.x * 256 + threadIdx.x;
  if (i < n) out[i] = (__bf16)in[i];
}

// Block tile 128x128, 256 threads = 8 waves, each wave 32(M) x 64(N)
// = 2x4 tiles of 16x16, accumulated in v8f (f32) via WMMA bf16.
template <int MODE>
__global__ __launch_bounds__(256)
void sgcn_gemm_bf16_wmma(
    const __bf16* __restrict__ A, const __bf16* __restrict__ B,
    int M, int N, int K, long sA, long sB, long sC,
    const float* __restrict__ bias,
    const float* __restrict__ bn_g, const float* __restrict__ bn_b,
    const float* __restrict__ bn_m, const float* __restrict__ bn_v,
    const __bf16* __restrict__ addsrc,
    __bf16* __restrict__ outb, float* __restrict__ outf)
{
  __shared__ __bf16 lA[2][128 * 32];   // [m][k], async-filled
  __shared__ __bf16 lB[2][128 * 32];   // [n][k]  (transposed on the way in)

  const int z = blockIdx.z;
  A += (long)z * sA;
  B += (long)z * sB;
  if constexpr ((MODE & EP_ADDRELU) != 0) addsrc += (long)z * sC;
  if constexpr ((MODE & EP_F32OUT) != 0) outf += (long)z * sC;
  else                                    outb += (long)z * sC;

  const int bm   = blockIdx.y * 128;
  const int bn   = blockIdx.x * 128;
  const int tid  = threadIdx.x;
  const int lane = tid & 31;
  const int wave = tid >> 5;
  const int wm   = (wave & 3) * 32;   // wave M offset inside block tile
  const int wn   = (wave >> 2) * 64;  // wave N offset inside block tile
  const int lm   = lane & 15;
  const int hi   = lane >> 4;

  // Async copy of A tile [128 m][32 k] into lA[buf]: 512 x 16B chunks,
  // 2 per thread, global -> LDS directly (no VGPR round trip).
  auto issueA = [&](int k0, int buf) {
#pragma unroll
    for (int i = 0; i < 2; ++i) {
      int idx = tid + i * 256;          // 0..511
      int row = idx >> 2;               // 4 chunks of 8 elems per row
      int c8  = (idx & 3) * 8;
      const __bf16* g = A + (long)(bm + row) * K + k0 + c8;
      unsigned lds = (unsigned)(unsigned long long)&lA[buf][row * 32 + c8];
      asm volatile("global_load_async_to_lds_b128 %0, %1, off"
                   :: "v"(lds), "v"((unsigned long long)g) : "memory");
    }
  };
  // Register-prefetch B tile [32 k][128 n] (16B x2 per thread).
  auto loadB = [&](int k0, v8bf* r) {
#pragma unroll
    for (int i = 0; i < 2; ++i) {
      int idx = tid + i * 256;
      int kr  = idx >> 4;               // 16 chunks of 8 per k-row
      int c8  = (idx & 15) * 8;
      r[i] = *(const v8bf*)(B + (long)(k0 + kr) * N + bn + c8);
    }
  };
  // Transposed store of prefetched B regs -> lB[buf] as [n][k].
  auto storeB = [&](const v8bf* r, int buf) {
#pragma unroll
    for (int i = 0; i < 2; ++i) {
      int idx = tid + i * 256;
      int kr  = idx >> 4;
      int c8  = (idx & 15) * 8;
#pragma unroll
      for (int e = 0; e < 8; ++e) lB[buf][(c8 + e) * 32 + kr] = r[i][e];
    }
  };

  v8f acc[2][4] = {};
  v8bf breg[2];

  const int T = K >> 5;                 // K / 32 steps
  loadB(0, breg);
  issueA(0, 0);

  for (int t = 0; t < T; ++t) {
    const int cur = t & 1;
    // finish staging tile t: B regs -> LDS (transposed), A async landed.
    storeB(breg, cur);
    asm volatile("s_wait_asynccnt 0" ::: "memory");
    __syncthreads();

    // kick off tile t+1 while tile t computes
    if (t + 1 < T) {
      loadB((t + 1) * 32, breg);
      issueA((t + 1) * 32, 1 - cur);
      if (t + 2 < T)  // warm GL2 for the t+2 B slab (global_prefetch_b8)
        __builtin_prefetch(B + (long)((t + 2) * 32 + (tid >> 4)) * N + bn + (tid & 15) * 8, 0, 0);
    }

    // --- fragments ---
    // A 16x32 bf16: lane<16 -> M=lane, K=0..7 & 16..23 ; lane>=16 -> K=8..15 & 24..31
    v16bf af[2], bfm[4];
#pragma unroll
    for (int i = 0; i < 2; ++i) {
      const __bf16* p = &lA[cur][(wm + i * 16 + lm) * 32 + hi * 8];
      union { v16bf v; v8bf h[2]; } u;
      u.h[0] = *(const v8bf*)p;
      u.h[1] = *(const v8bf*)(p + 16);
      af[i] = u.v;
    }
    // B 32x16 bf16: n = lane&15, K = (lane>>4)*8 + 0..7 & 16 + (lane>>4)*8 + 0..7 (LDS is [n][k])
#pragma unroll
    for (int j = 0; j < 4; ++j) {
      const __bf16* p = &lB[cur][(wn + j * 16 + lm) * 32 + hi * 8];
      union { v16bf v; v8bf h[2]; } u;
      u.h[0] = *(const v8bf*)p;
      u.h[1] = *(const v8bf*)(p + 16);
      bfm[j] = u.v;
    }
#pragma unroll
    for (int i = 0; i < 2; ++i)
#pragma unroll
      for (int j = 0; j < 4; ++j)
        acc[i][j] = __builtin_amdgcn_wmma_f32_16x16x32_bf16(
            false, af[i], false, bfm[j], (short)0, acc[i][j], false, false);
    // no trailing barrier needed: next iteration's LDS writes go to the
    // other buffer, and its barrier orders buffer reuse (lockstep window).
  }

  // --- epilogue: C/D 16x16 f32 layout: elem r of lane l -> M=r+(l>>4)*8, N=l&15
#pragma unroll
  for (int i = 0; i < 2; ++i) {
#pragma unroll
    for (int j = 0; j < 4; ++j) {
#pragma unroll
      for (int r = 0; r < 8; ++r) {
        int grow = bm + wm + i * 16 + hi * 8 + r;
        int gcol = bn + wn + j * 16 + lm;
        float c = acc[i][j][r];
        if constexpr ((MODE & EP_BIAS) != 0) c += bias[grow];
        if constexpr ((MODE & EP_BN) != 0) {
          float sc = bn_g[grow] * rsqrtf(bn_v[grow] + 1e-5f);
          c = (c - bn_m[grow]) * sc + bn_b[grow];
        }
        if constexpr ((MODE & EP_ADDRELU) != 0) {
          c += (float)addsrc[(long)grow * N + gcol];
          c = fmaxf(c, 0.0f);
        }
        if constexpr ((MODE & EP_F32OUT) != 0) {
          if constexpr ((MODE & EP_TRANS) != 0) outf[(long)gcol * M + grow] = c;
          else                                  outf[(long)grow * N + gcol] = c;
        } else {
          if constexpr ((MODE & EP_TRANS) != 0) outb[(long)gcol * M + grow] = (__bf16)c;
          else                                  outb[(long)grow * N + gcol] = (__bf16)c;
        }
      }
    }
  }
}

// softmax over axis 2 of S[b][p][q], write transposed A^T[b][q][p] in bf16
__global__ __launch_bounds__(256)
void sgcn_softmax_T(const float* __restrict__ S, __bf16* __restrict__ At, int P) {
  __shared__ float red[256];
  const int b = blockIdx.y, p = blockIdx.x, q = threadIdx.x;
  const float* row = S + ((long)b * P + p) * P;
  float x = row[q];
  red[q] = x; __syncthreads();
  for (int s = 128; s > 0; s >>= 1) {
    if (q < s) red[q] = fmaxf(red[q], red[q + s]);
    __syncthreads();
  }
  float mx = red[0]; __syncthreads();
  float e = __expf(x - mx);
  red[q] = e; __syncthreads();
  for (int s = 128; s > 0; s >>= 1) {
    if (q < s) red[q] += red[q + s];
    __syncthreads();
  }
  float inv = 1.0f / red[0];
  At[((long)b * P + q) * P + p] = (__bf16)(e * inv);
}

extern "C" void kernel_launch(void* const* d_in, const int* in_sizes, int n_in,
                              void* d_out, int out_size, void* d_ws, size_t ws_size,
                              hipStream_t stream) {
  constexpr int Bn = 8, P = 256, C2 = 512, N = 4096;  // N = 64*64

  const float* x    = (const float*)d_in[0];
  const float* y    = (const float*)d_in[1];
  const float* wv   = (const float*)d_in[2];
  const float* bv   = (const float*)d_in[3];
  const float* wq   = (const float*)d_in[4];
  const float* bq   = (const float*)d_in[5];
  const float* w1d  = (const float*)d_in[6];
  const float* g1   = (const float*)d_in[7];
  const float* b1   = (const float*)d_in[8];
  const float* m1   = (const float*)d_in[9];
  const float* v1   = (const float*)d_in[10];
  const float* wo   = (const float*)d_in[11];
  const float* bo   = (const float*)d_in[12];
  const float* go   = (const float*)d_in[13];
  const float* bbo  = (const float*)d_in[14];
  const float* mo   = (const float*)d_in[15];
  const float* vo   = (const float*)d_in[16];
  const float* wx1  = (const float*)d_in[17];
  const float* bx1  = (const float*)d_in[18];
  const float* gx1  = (const float*)d_in[19];
  const float* bbx1 = (const float*)d_in[20];
  const float* mx1  = (const float*)d_in[21];
  const float* vx1  = (const float*)d_in[22];
  const float* wx2  = (const float*)d_in[23];
  const float* bx2  = (const float*)d_in[24];
  const float* gx2  = (const float*)d_in[25];
  const float* bbx2 = (const float*)d_in[26];
  const float* mx2  = (const float*)d_in[27];
  const float* vx2  = (const float*)d_in[28];

  char* ws = (char*)d_ws;
  size_t off = 0;
  auto carve = [&](size_t bytes) -> void* {
    void* p = ws + off;
    off += bytes;
    off = (off + 255) & ~(size_t)255;
    return p;
  };

  __bf16* xb   = (__bf16*)carve((size_t)Bn * C2 * N * 2);  // x bf16      [b][512][4096]
  __bf16* yb   = (__bf16*)carve((size_t)Bn * P * N * 2);   // y bf16      [b][256][4096]
  __bf16* wvb  = (__bf16*)carve((size_t)P * C2 * 2);
  __bf16* wqb  = (__bf16*)carve((size_t)P * C2 * 2);
  __bf16* w1db = (__bf16*)carve((size_t)P * P * 2);
  __bf16* wob  = (__bf16*)carve((size_t)P * P * 2);
  __bf16* wx1b = (__bf16*)carve((size_t)P * C2 * 2);
  __bf16* wx2b = (__bf16*)carve((size_t)P * P * 2);
  __bf16* vT   = (__bf16*)carve((size_t)Bn * N * P * 2);   // node_v^T    [b][4096][256]
  __bf16* qm   = (__bf16*)carve((size_t)Bn * P * N * 2);   // node_q      [b][256][4096]
  float*  S    = (float*) carve((size_t)Bn * P * P * 4);   // q @ v       [b][256][256]
  __bf16* At   = (__bf16*)carve((size_t)Bn * P * P * 2);   // softmax^T   [b][256][256]
  __bf16* av   = (__bf16*)carve((size_t)Bn * P * N * 2);   // AV          [b][256][4096]
  __bf16* avw  = (__bf16*)carve((size_t)Bn * P * N * 2);   // AVW+bn      [b][256][4096]
  __bf16* dec  = (__bf16*)carve((size_t)Bn * P * N * 2);   // out_dec     [b][256][4096]
  __bf16* xp1  = (__bf16*)carve((size_t)Bn * P * N * 2);   // xpre1       [b][256][4096]

  // ---- precision cast (f32 -> bf16) ----
  auto cvt = [&](const float* src, __bf16* dst, int n) {
    sgcn_cvt_f32_bf16<<<(n + 255) / 256, 256, 0, stream>>>(src, dst, n);
  };
  cvt(x,   xb,   Bn * C2 * N);
  cvt(y,   yb,   Bn * P * N);
  cvt(wv,  wvb,  P * C2);
  cvt(wq,  wqb,  P * C2);
  cvt(w1d, w1db, P * P);
  cvt(wo,  wob,  P * P);
  cvt(wx1, wx1b, P * C2);
  cvt(wx2, wx2b, P * P);

  const long sX  = (long)C2 * N;   // batch stride of x
  const long sPN = (long)P * N;    // batch stride of [256][4096] tensors
  const long sPP = (long)P * P;

  dim3 gWide(N / 128, P / 128, Bn);   // (32, 2, 8)
  dim3 gSq(P / 128, P / 128, Bn);     // (2, 2, 8)

  // 1) node_v^T[b][n][o] = (wv @ x[b])^T + bv   (transposed bf16 store)
  sgcn_gemm_bf16_wmma<EP_BIAS | EP_TRANS><<<gWide, 256, 0, stream>>>(
      wvb, xb, P, N, C2, 0, sX, sPN, bv,
      nullptr, nullptr, nullptr, nullptr, nullptr, vT, nullptr);

  // 2) node_q[b] = wq @ x[b] + bq
  sgcn_gemm_bf16_wmma<EP_BIAS><<<gWide, 256, 0, stream>>>(
      wqb, xb, P, N, C2, 0, sX, sPN, bq,
      nullptr, nullptr, nullptr, nullptr, nullptr, qm, nullptr);

  // 3) S[b] = node_q @ node_v  (f32, K = 4096)
  sgcn_gemm_bf16_wmma<EP_F32OUT><<<gSq, 256, 0, stream>>>(
      qm, vT, P, P, N, sPN, sPN, sPP, nullptr,
      nullptr, nullptr, nullptr, nullptr, nullptr, nullptr, S);

  // 4) A^T = softmax(S, axis=2)^T
  sgcn_softmax_T<<<dim3(P, Bn), 256, 0, stream>>>(S, At, P);

  // 5) AV[b] = A^T @ y[b]
  sgcn_gemm_bf16_wmma<0><<<gWide, 256, 0, stream>>>(
      At, yb, P, N, P, sPP, sPN, sPN, nullptr,
      nullptr, nullptr, nullptr, nullptr, nullptr, av, nullptr);

  // 6) AVW = BN1d(w1d @ AV)
  sgcn_gemm_bf16_wmma<EP_BN><<<gWide, 256, 0, stream>>>(
      w1db, av, P, N, P, 0, sPN, sPN, nullptr,
      g1, b1, m1, v1, nullptr, avw, nullptr);

  // 7) out_dec = BN(wo @ AVW + bo)
  sgcn_gemm_bf16_wmma<EP_BIAS | EP_BN><<<gWide, 256, 0, stream>>>(
      wob, avw, P, N, P, 0, sPN, sPN, bo,
      go, bbo, mo, vo, nullptr, dec, nullptr);

  // 8) xp1 = BN(wx1 @ x + bx1)
  sgcn_gemm_bf16_wmma<EP_BIAS | EP_BN><<<gWide, 256, 0, stream>>>(
      wx1b, xb, P, N, C2, 0, sX, sPN, bx1,
      gx1, bbx1, mx1, vx1, nullptr, xp1, nullptr);

  // 9) out = relu(BN(wx2 @ xp1 + bx2) + out_dec)  (f32 store to d_out)
  sgcn_gemm_bf16_wmma<EP_BIAS | EP_BN | EP_ADDRELU | EP_F32OUT><<<gWide, 256, 0, stream>>>(
      wx2b, xp1, P, N, P, 0, sPN, sPN, bx2,
      gx2, bbx2, mx2, vx2, dec, nullptr, (float*)d_out);
}
